// Model_25451976196117
// MI455X (gfx1250) — matrix operand order repaired and checked by reading, not compiled, not run on hardware
//
#include <hip/hip_runtime.h>
#include <math.h>

// ---------- CDNA5 WMMA types ----------
typedef __attribute__((ext_vector_type(16))) __bf16       v16bf;
typedef __attribute__((ext_vector_type(8)))  float        v8f;
typedef __attribute__((ext_vector_type(4)))  unsigned int u32x4;

union Frag { u32x4 q[2]; v16bf v; };   // 32 bytes: 16 bf16 elems per lane

#define WMMA_BF16 __builtin_amdgcn_wmma_f32_16x16x32_bf16

__device__ __forceinline__ unsigned short f2bf(float x) {
  unsigned int u = __float_as_uint(x);
  u += 0x7FFFu + ((u >> 16) & 1u);          // round-to-nearest-even
  return (unsigned short)(u >> 16);
}

// Problem constants: B=64 T=512 E=768 H=256 L=9, G=4H=1024, M=T*B=32768

// ---------- 0) dtype conversions ----------
__global__ void cvt_emb_kernel(const float* __restrict__ in, unsigned short* __restrict__ out) {
  // out[(t*64+b)*768+e] = bf16(in[(b*512+t)*768+e])   (does the (B,T)->(T,B) transpose)
  const long idx = (long)blockIdx.x * 256 + threadIdx.x;     // exactly 25,165,824 threads
  const int  e = (int)(idx % 768);
  const long m = idx / 768;
  const int  t = (int)(m / 64);
  const int  b = (int)(m % 64);
  out[idx] = f2bf(in[((long)b * 512 + t) * 768 + e]);
}

__global__ void cvt_w_kernel(const float* __restrict__ in, unsigned short* __restrict__ out, int n) {
  int i = blockIdx.x * 256 + threadIdx.x;
  if (i < n) out[i] = f2bf(in[i]);
}

// ---------- 1) input projection GEMM: xg = x @ W_ihT + b  (both directions) ----------
// M=32768, N=1024, K=768. Block = 8 waves = 64M x 128N; wave = 16M x 64N (4 accum tiles).
// Software-pipelined: fragments for k+32 are loaded while the k WMMAs execute.
__global__ __launch_bounds__(256) void proj_gemm_kernel(
    const unsigned short* __restrict__ X,    // [32768][768] bf16 row-major
    const unsigned short* __restrict__ Wf,   // [1024][768] bf16 row-major (g,e)
    const unsigned short* __restrict__ Wb,
    const float* __restrict__ biasf, const float* __restrict__ biasb,
    float* __restrict__ outf, float* __restrict__ outb)
{
  const int dir = blockIdx.z;
  const unsigned short* W   = dir ? Wb    : Wf;
  const float*          bia = dir ? biasb : biasf;
  float*                out = dir ? outb  : outf;

  const int lane = threadIdx.x & 31;
  const int wave = threadIdx.x >> 5;
  const int wm = wave >> 1, wn = wave & 1;
  const int m0 = blockIdx.y * 64 + wm * 16;
  const int nb = blockIdx.x * 128 + wn * 64;

  const int l15 = lane & 15;
  const int hi  = lane >> 4;

  const unsigned short* xrow = X + (size_t)(m0 + l15) * 768 + (hi << 3);   // A: +k
  const unsigned short* wrow[4];
  #pragma unroll
  for (int tt = 0; tt < 4; ++tt)
    wrow[tt] = W + (size_t)(nb + tt * 16 + l15) * 768 + (hi << 3);         // B: +k

  v8f zero = {};
  v8f acc[4];
  #pragma unroll
  for (int tt = 0; tt < 4; ++tt) acc[tt] = zero;

  Frag a[2], bq[2][4];
  a[0].q[0] = *(const u32x4*)(xrow);
  a[0].q[1] = *(const u32x4*)(xrow + 16);
  #pragma unroll
  for (int tt = 0; tt < 4; ++tt) {
    bq[0][tt].q[0] = *(const u32x4*)(wrow[tt]);
    bq[0][tt].q[1] = *(const u32x4*)(wrow[tt] + 16);
  }

  for (int kk = 0; kk < 24; ++kk) {
    const int cur = kk & 1, nxt = cur ^ 1;
    if (kk + 1 < 24) {                       // prefetch k+32 while computing k
      const int k2 = (kk + 1) * 32;
      a[nxt].q[0] = *(const u32x4*)(xrow + k2);
      a[nxt].q[1] = *(const u32x4*)(xrow + k2 + 16);
      #pragma unroll
      for (int tt = 0; tt < 4; ++tt) {
        bq[nxt][tt].q[0] = *(const u32x4*)(wrow[tt] + k2);
        bq[nxt][tt].q[1] = *(const u32x4*)(wrow[tt] + k2 + 16);
      }
    }
    #pragma unroll
    for (int tt = 0; tt < 4; ++tt)
      acc[tt] = WMMA_BF16(false, a[cur].v, false, bq[cur][tt].v, (short)0, acc[tt], false, false);
  }

  // C/D layout: n = lane&15, m = vgpr + 8*(lane>=16). NT stores: 268 MB stream, bypass-ish L2.
  #pragma unroll
  for (int tt = 0; tt < 4; ++tt) {
    const int n = nb + tt * 16 + l15;
    const float bv = bia[n];
    #pragma unroll
    for (int v = 0; v < 8; ++v) {
      const int m = m0 + v + (hi << 3);
      __builtin_nontemporal_store(acc[tt][v] + bv, &out[(size_t)m * 1024 + n]);
    }
  }
}

// ---------- 2) LSTM recurrence (persistent, 8 blocks: 2 dirs x 4 batch-groups of 16) ----------
// Per step: gates[16][1024] = xg[t](LDS, async-prefetched) + h_bf16 @ W_hhT (WMMA), then cell.
// LDS layout (byte offsets into dynamic smem):
#define HSH_OFF   0u          // ushort[16][264]  =   8448 B (pad 8 -> conflict-free b128 reads)
#define CST_OFF   8448u       // float [16][256]  =  16384 B
#define GAT_OFF   24832u      // float [16][1032] =  66048 B
#define XBUF_OFF  90880u      // float [2][16][1024] = 131072 B (async double buffer)
#define LDS_TOTAL 221952u

__global__ __launch_bounds__(256) void lstm_kernel(
    const float* __restrict__ xg_f, const float* __restrict__ xg_b,
    const unsigned short* __restrict__ whh_f, const unsigned short* __restrict__ whh_b,
    float* __restrict__ feats)                 // [512][64][512] f32, fwd cols 0..255, bwd 256..511
{
  extern __shared__ char smem[];
  unsigned short* hsh = (unsigned short*)(smem + HSH_OFF);
  float* cst  = (float*)(smem + CST_OFF);
  float* gat  = (float*)(smem + GAT_OFF);
  float* xbuf = (float*)(smem + XBUF_OFF);
  const unsigned smem_base = (unsigned)(uintptr_t)smem;   // LDS byte offset (flat addr truncates)

  const int dir = blockIdx.x >> 2;
  const int b0  = (blockIdx.x & 3) * 16;
  const float* xg          = dir ? xg_b  : xg_f;
  const unsigned short* W  = dir ? whh_b : whh_f;

  const int tid  = threadIdx.x;
  const int lane = tid & 31;
  const int wave = tid >> 5;
  const int hi   = lane >> 4;
  const int l15  = lane & 15;
  const int nbase = wave * 128;      // each wave owns 128 gate columns (8 N-tiles)

  for (int i = tid; i < 16 * 264; i += 256) hsh[i] = 0;
  for (int i = tid; i < 16 * 256; i += 256) cst[i] = 0.f;
  __syncthreads();

  // Each wave async-loads exactly the xg slice it consumes: rows 0..15, cols [nbase,nbase+128).
  // 8 KB/wave = 16 x b128 per lane-set; no cross-wave dependency -> own ASYNCcnt wait suffices.
  auto issue_xg = [&](int t_, int buf_) {
    const unsigned long long gbase =
        (unsigned long long)(uintptr_t)(xg + ((size_t)t_ * 64 + b0) * 1024);
    const unsigned lbase = smem_base + XBUF_OFF + (unsigned)buf_ * 65536u;
    #pragma unroll
    for (int i = 0; i < 16; ++i) {
      const int c = lane + 32 * i;                          // 0..511 16B-chunks
      const unsigned goff =
          (unsigned)((((c >> 5) * 1024) + nbase + ((c & 31) << 2)) * 4);
      asm volatile("global_load_async_to_lds_b128 %0, %1, %2"
                   :: "v"(lbase + goff), "v"(goff), "s"(gbase) : "memory");
    }
  };

  issue_xg(dir ? 511 : 0, 0);       // prologue: first step's tile into buf0

  v8f zero = {};
  for (int s = 0; s < 512; ++s) {
    const int t = dir ? (511 - s) : s;

    if (s + 1 < 512) issue_xg(dir ? (510 - s) : (s + 1), (s + 1) & 1);

    // --- load all 8 A-fragments of h (16x256 bf16) from LDS once per step ---
    Frag a[8];
    #pragma unroll
    for (int kk = 0; kk < 8; ++kk) {
      const unsigned short* hp = hsh + l15 * 264 + kk * 32 + (hi << 3);
      a[kk].q[0] = *(const u32x4*)(hp);          // ds_read_b128
      a[kk].q[1] = *(const u32x4*)(hp + 16);
    }

    // current buffer must be complete; next step's 16 loads may stay in flight
    if (s + 1 < 512) asm volatile("s_wait_asynccnt 0x10" ::: "memory");
    else             asm volatile("s_wait_asynccnt 0x0"  ::: "memory");
    const float* xls = xbuf + ((size_t)(s & 1) << 14);

    // --- GEMM: two interleaved WMMA accumulation chains per tile-pair ---
    #pragma unroll
    for (int tp = 0; tp < 4; ++tp) {
      const int n0 = nbase + (2 * tp) * 16 + l15;
      const int n1 = n0 + 16;
      const unsigned short* wp0 = W + (size_t)n0 * 256 + (hi << 3);
      const unsigned short* wp1 = W + (size_t)n1 * 256 + (hi << 3);
      v8f acc0 = zero, acc1 = zero;
      #pragma unroll
      for (int kk = 0; kk < 8; ++kk) {
        Frag q0, q1;
        q0.q[0] = *(const u32x4*)(wp0 + kk * 32);
        q0.q[1] = *(const u32x4*)(wp0 + kk * 32 + 16);
        q1.q[0] = *(const u32x4*)(wp1 + kk * 32);
        q1.q[1] = *(const u32x4*)(wp1 + kk * 32 + 16);
        acc0 = WMMA_BF16(false, a[kk].v, false, q0.v, (short)0, acc0, false, false);
        acc1 = WMMA_BF16(false, a[kk].v, false, q1.v, (short)0, acc1, false, false);
      }
      #pragma unroll
      for (int v = 0; v < 8; ++v) {
        const int m = v + (hi << 3);
        gat[m * 1032 + n0] = acc0[v] + xls[m * 1024 + n0];
        gat[m * 1032 + n1] = acc1[v] + xls[m * 1024 + n1];
      }
    }
    __syncthreads();

    // --- cell: i,f,g,o split across the 1024 gate columns ---
    #pragma unroll 4
    for (int r = 0; r < 16; ++r) {
      const int col = tid;               // 256 threads = 256 hidden units
      const float gi = gat[r * 1032 + col];
      const float gf = gat[r * 1032 + col + 256];
      const float gg = gat[r * 1032 + col + 512];
      const float go = gat[r * 1032 + col + 768];
      const float si = 1.f / (1.f + __expf(-gi));
      const float sf = 1.f / (1.f + __expf(-gf));
      const float so = 1.f / (1.f + __expf(-go));
      const float cn = sf * cst[r * 256 + col] + si * tanhf(gg);
      const float hn = so * tanhf(cn);
      cst[r * 256 + col] = cn;
      hsh[r * 264 + col] = f2bf(hn);
      feats[((size_t)t * 64 + b0 + r) * 512 + dir * 256 + col] = hn;
    }
    __syncthreads();
  }
}

// ---------- 3) emission scores: em = feats @ W_linT + b_lin  (L=9) ----------
__global__ __launch_bounds__(256) void emis_kernel(
    const float* __restrict__ feats, const float* __restrict__ Wlin,
    const float* __restrict__ blin, float* __restrict__ em)
{
  const int wave = threadIdx.x >> 5;
  const int lane = threadIdx.x & 31;
  const long r = (long)blockIdx.x * 8 + wave;     // row = t*64+b, 0..32767
  float f[16];
  #pragma unroll
  for (int j = 0; j < 16; ++j) f[j] = feats[r * 512 + lane + j * 32];
  #pragma unroll
  for (int l = 0; l < 9; ++l) {
    float sacc = 0.f;
    #pragma unroll
    for (int j = 0; j < 16; ++j) sacc += f[j] * Wlin[l * 512 + lane + j * 32];
    #pragma unroll
    for (int d = 16; d >= 1; d >>= 1) sacc += __shfl_xor(sacc, d, 32);
    if (lane == 0) em[r * 9 + l] = sacc + blin[l];
  }
}

// ---------- 4) CRF numerator + forward algorithm, one wave per batch element ----------
__global__ __launch_bounds__(32) void crf_kernel(
    const float* __restrict__ em, const float* __restrict__ start_trans,
    const float* __restrict__ end_trans, const float* __restrict__ trans,
    const int* __restrict__ mask, const int* __restrict__ labels,
    float* __restrict__ perb)
{
  __shared__ float st[81];
  const int b = blockIdx.x;
  const int lane = threadIdx.x;
  for (int i = lane; i < 81; i += 32) st[i] = trans[i];
  __syncthreads();

  // numerator (gold path score), lanes stride over t
  float accn = 0.f; int cnt = 0;
  for (int t = lane; t < 512; t += 32) {
    const int mt = mask[b * 512 + t];
    cnt += (mt != 0);
    if (t >= 1 && mt) {
      const int tp = labels[b * 512 + t - 1], tc = labels[b * 512 + t];
      accn += st[tp * 9 + tc] + em[((size_t)t * 64 + b) * 9 + tc];
    }
  }
  #pragma unroll
  for (int d = 16; d >= 1; d >>= 1) { accn += __shfl_xor(accn, d, 32); cnt += __shfl_xor(cnt, d, 32); }
  const int tag0 = labels[b * 512];
  float num = start_trans[tag0] + em[(size_t)b * 9 + tag0] + accn;
  num += end_trans[labels[b * 512 + (cnt - 1)]];

  // partition function: lane j<9 holds score[j]
  const int j = lane;
  float score = (j < 9) ? (start_trans[j] + em[(size_t)b * 9 + j]) : -1e30f;
  for (int t = 1; t < 512; ++t) {
    const float emt = (j < 9) ? em[((size_t)t * 64 + b) * 9 + j] : 0.f;
    float vs[9], mx = -1e30f;
    #pragma unroll
    for (int i = 0; i < 9; ++i) {
      const float si = __shfl(score, i, 32);
      const float v = si + ((j < 9) ? st[i * 9 + j] : 0.f);
      vs[i] = v; mx = fmaxf(mx, v);
    }
    float sum = 0.f;
    #pragma unroll
    for (int i = 0; i < 9; ++i) sum += __expf(vs[i] - mx);
    const float nxt = emt + mx + logf(sum);
    if (mask[b * 512 + t]) score = (j < 9) ? nxt : score;
  }
  float v = (j < 9) ? (score + end_trans[j]) : -1e30f;
  float mx = v;
  #pragma unroll
  for (int d = 16; d >= 1; d >>= 1) mx = fmaxf(mx, __shfl_xor(mx, d, 32));
  float sum = (j < 9) ? __expf(v - mx) : 0.f;
  #pragma unroll
  for (int d = 16; d >= 1; d >>= 1) sum += __shfl_xor(sum, d, 32);
  if (lane == 0) perb[b] = (mx + logf(sum)) - num;
}

// ---------- 5) mean over batch ----------
__global__ void reduce_kernel(const float* __restrict__ perb, float* __restrict__ out) {
  __shared__ float sm[64];
  const int tid = threadIdx.x;
  sm[tid] = perb[tid];
  __syncthreads();
  for (int s = 32; s > 0; s >>= 1) { if (tid < s) sm[tid] += sm[tid + s]; __syncthreads(); }
  if (tid == 0) out[0] = sm[0] * (1.f / 64.f);
}

// ---------- host ----------
extern "C" void kernel_launch(void* const* d_in, const int* in_sizes, int n_in,
                              void* d_out, int out_size, void* d_ws, size_t ws_size,
                              hipStream_t stream)
{
  (void)in_sizes; (void)n_in; (void)out_size; (void)ws_size;
  const float* emb        = (const float*)d_in[0];
  const float* W_ih_f     = (const float*)d_in[1];
  const float* W_hh_f     = (const float*)d_in[2];
  const float* b_f        = (const float*)d_in[3];
  const float* W_ih_b     = (const float*)d_in[4];
  const float* W_hh_b     = (const float*)d_in[5];
  const float* b_b        = (const float*)d_in[6];
  const float* W_lin      = (const float*)d_in[7];
  const float* b_lin      = (const float*)d_in[8];
  const float* start_tr   = (const float*)d_in[9];
  const float* end_tr     = (const float*)d_in[10];
  const float* trans      = (const float*)d_in[11];
  const int*   mask       = (const int*)d_in[12];
  const int*   labels     = (const int*)d_in[13];

  char* ws = (char*)d_ws;
  size_t off = 0;
  auto take = [&](size_t bytes) -> char* {
    char* p = ws + off;
    off = (off + bytes + 255) & ~(size_t)255;
    return p;
  };
  unsigned short* embbf = (unsigned short*)take((size_t)32768 * 768 * 2);
  unsigned short* wihf  = (unsigned short*)take((size_t)1024 * 768 * 2);
  unsigned short* wihb  = (unsigned short*)take((size_t)1024 * 768 * 2);
  unsigned short* whhf  = (unsigned short*)take((size_t)1024 * 256 * 2);
  unsigned short* whhb  = (unsigned short*)take((size_t)1024 * 256 * 2);
  float* xgf   = (float*)take((size_t)32768 * 1024 * 4);
  float* xgb   = (float*)take((size_t)32768 * 1024 * 4);
  float* feats = (float*)take((size_t)32768 * 512 * 4);
  float* em    = (float*)take((size_t)32768 * 9 * 4);
  float* perb  = (float*)take((size_t)64 * 4);

  cvt_emb_kernel<<<98304, 256, 0, stream>>>(emb, embbf);
  cvt_w_kernel<<<(1024 * 768 + 255) / 256, 256, 0, stream>>>(W_ih_f, wihf, 1024 * 768);
  cvt_w_kernel<<<(1024 * 768 + 255) / 256, 256, 0, stream>>>(W_ih_b, wihb, 1024 * 768);
  cvt_w_kernel<<<(1024 * 256 + 255) / 256, 256, 0, stream>>>(W_hh_f, whhf, 1024 * 256);
  cvt_w_kernel<<<(1024 * 256 + 255) / 256, 256, 0, stream>>>(W_hh_b, whhb, 1024 * 256);

  proj_gemm_kernel<<<dim3(8, 512, 2), 256, 0, stream>>>(embbf, wihf, wihb, b_f, b_b, xgf, xgb);

  lstm_kernel<<<8, 256, LDS_TOTAL, stream>>>(xgf, xgb, whhf, whhb, feats);

  emis_kernel<<<4096, 256, 0, stream>>>(feats, W_lin, b_lin, em);
  crf_kernel<<<64, 32, 0, stream>>>(em, start_tr, end_tr, trans, mask, labels, perb);
  reduce_kernel<<<1, 64, 0, stream>>>(perb, (float*)d_out);
}
